// SimpleRNN_9947144258266
// MI455X (gfx1250) — matrix operand order repaired and checked by reading, not compiled, not run on hardware
//
#include <hip/hip_runtime.h>
#include <hip/hip_bf16.h>
#include <math.h>

// Problem constants (match reference)
#define B_ 64
#define T_ 2048
#define V_ 32000
#define W_ 256
#define E_ 512
#define C_ 2

typedef __attribute__((ext_vector_type(8)))  float  v8f;
typedef __attribute__((ext_vector_type(8)))  float  f8;
typedef __attribute__((ext_vector_type(8)))  __bf16 v8bf;
typedef __attribute__((ext_vector_type(16))) __bf16 v16bf;

static __device__ __forceinline__ __bf16 f2bf(float f) {
  union { float f; unsigned u; } v; v.f = f;
  unsigned r = v.u + 0x7FFFu + ((v.u >> 16) & 1u);   // round-to-nearest-even
  unsigned short h = (unsigned short)(r >> 16);
  return __builtin_bit_cast(__bf16, h);
}

static __device__ __forceinline__ float fast_tanh(float x) {
#if __has_builtin(__builtin_amdgcn_tanhf)
  return __builtin_amdgcn_tanhf(x);      // v_tanh_f32 (gfx1250 TRANS unit)
#else
  return tanhf(x);
#endif
}

// ---------------------------------------------------------------------------
// Convert fp32 [K][N] row-major -> bf16 [N][K] (B-fragment friendly layout)
// ---------------------------------------------------------------------------
__global__ void cvt_transpose(const float* __restrict__ in,
                              __bf16* __restrict__ out, int K, int N) {
  int i = blockIdx.x * blockDim.x + threadIdx.x;
  if (i >= K * N) return;
  int k = i / N, n = i % N;
  out[(size_t)n * K + k] = f2bf(in[i]);
}

// ---------------------------------------------------------------------------
// xp1 = emb[x] @ Wx1 + b1,  TIME-MAJOR output: row m = t*B + b.
// One wave computes a 16x64 strip (4 WMMA accumulators); 8 waves / block.
// A rows are gathered via the token index array and converted fp32->bf16.
// ---------------------------------------------------------------------------
__global__ __launch_bounds__(256) void gemm_embed(
    const int* __restrict__ idx,     // [B][T] token ids
    const float* __restrict__ emb,   // [V][E] fp32
    const __bf16* __restrict__ WT,   // [N][K] bf16 (transposed)
    const float* __restrict__ bias,  // [N]
    float* __restrict__ out)         // [T*B][N] fp32, m = t*B + b
{
  const int lane  = threadIdx.x & 31;
  const int wave  = threadIdx.x >> 5;
  const int strip = blockIdx.x * 8 + wave;     // (M/16)*(N/64) strips
  const int mt    = strip >> 2;
  const int n0    = (strip & 3) * 64;
  const int r     = lane & 15;
  const int half  = lane >> 4;

  const int m = mt * 16 + r;                    // time-major row: t*B + b
  const int b = m & (B_ - 1);
  const int t = m >> 6;                         // B_ == 64
  const float* arow = emb + (size_t)idx[b * T_ + t] * E_;

  v8f acc[4] = {};
  for (int kb = 0; kb < E_ / 32; ++kb) {
    const int ka = kb * 32 + half * 16;
    f8 f0 = *(const f8*)(arow + ka);            // K = ka..ka+7
    f8 f1 = *(const f8*)(arow + ka + 8);        // K = ka+8..ka+15
    v16bf a;
#pragma unroll
    for (int i = 0; i < 8; ++i) { a[i] = f2bf(f0[i]); a[i + 8] = f2bf(f1[i]); }
#pragma unroll
    for (int j = 0; j < 4; ++j) {
      const __bf16* bp = WT + (size_t)(n0 + j * 16 + r) * E_ + kb * 32 + half * 16;
      v16bf bb = *(const v16bf*)bp;             // 16 contiguous K values
      acc[j] = __builtin_amdgcn_wmma_f32_16x16x32_bf16(
          false, a, false, bb, (short)0, acc[j], false, false);
    }
  }
#pragma unroll
  for (int j = 0; j < 4; ++j) {
    const int n = n0 + j * 16 + r;
    const float bn = bias[n];
#pragma unroll
    for (int i = 0; i < 8; ++i) {
      const int mm = mt * 16 + i + 8 * half;    // C layout: M = reg + 8*half
      out[(size_t)mm * W_ + n] = acc[j][i] + bn;
    }
  }
}

// ---------------------------------------------------------------------------
// xp2 = s1 @ Wx2 + b2  (A bf16 row-major, time-major rows; K=256)
// ---------------------------------------------------------------------------
__global__ __launch_bounds__(256) void gemm_bf16(
    const __bf16* __restrict__ A,    // [M][K] bf16
    const __bf16* __restrict__ WT,   // [N][K] bf16
    const float* __restrict__ bias,
    float* __restrict__ out, int K)
{
  const int lane  = threadIdx.x & 31;
  const int wave  = threadIdx.x >> 5;
  const int strip = blockIdx.x * 8 + wave;
  const int mt    = strip >> 2;
  const int n0    = (strip & 3) * 64;
  const int r     = lane & 15;
  const int half  = lane >> 4;

  const __bf16* arow = A + (size_t)(mt * 16 + r) * K;

  v8f acc[4] = {};
  for (int kb = 0; kb < K / 32; ++kb) {
    const int ka = kb * 32 + half * 16;
    v8bf l0 = *(const v8bf*)(arow + ka);
    v8bf l1 = *(const v8bf*)(arow + ka + 8);
    v16bf a;
#pragma unroll
    for (int i = 0; i < 8; ++i) { a[i] = l0[i]; a[i + 8] = l1[i]; }
#pragma unroll
    for (int j = 0; j < 4; ++j) {
      const __bf16* bp = WT + (size_t)(n0 + j * 16 + r) * K + kb * 32 + half * 16;
      v16bf bb = *(const v16bf*)bp;
      acc[j] = __builtin_amdgcn_wmma_f32_16x16x32_bf16(
          false, a, false, bb, (short)0, acc[j], false, false);
    }
  }
#pragma unroll
  for (int j = 0; j < 4; ++j) {
    const int n = n0 + j * 16 + r;
    const float bn = bias[n];
#pragma unroll
    for (int i = 0; i < 8; ++i) {
      const int mm = mt * 16 + i + 8 * half;
      out[(size_t)mm * W_ + n] = acc[j][i] + bn;
    }
  }
}

// ---------------------------------------------------------------------------
// Sequential RNN scan: h_t = tanh(xp_t + h_{t-1} @ Wh).
// Batch split across blocks (16 rows each) -> zero inter-block sync.
// Wh B-fragments resident in VGPRs; h double-buffered in LDS (16 KB);
// xp tile loaded into registers before the WMMA chain; explicit prefetch
// 4 steps ahead; one workgroup barrier per step.
// xp/seq layout is TIME-MAJOR: row m = t*B + b  (contiguous 64 KB per step).
// ---------------------------------------------------------------------------
template <bool WRITE_SEQ>
__global__ __launch_bounds__(512) void rnn_scan(
    const float* __restrict__ xp,    // [T*B][W] fp32
    const __bf16* __restrict__ WhT,  // [W][W] bf16 transposed
    __bf16* __restrict__ seq_out,    // [T*B][W] bf16 (layer 1 only)
    float* __restrict__ h_out)       // [B][W] fp32 final state (layer 2 only)
{
  __shared__ __bf16 Hb[2][16][W_];

  const int lane = threadIdx.x & 31;
  const int wave = threadIdx.x >> 5;      // 16 waves, one 16-col tile each
  const int r    = lane & 15;
  const int half = lane >> 4;
  const int n    = wave * 16 + r;         // output column for this lane
  const int b0   = blockIdx.x * 16;       // batch rows owned by this block

  // Resident B-matrix fragments: Wh^T[n][k], 16 contiguous K per fragment.
  v16bf bw[8];
#pragma unroll
  for (int kb = 0; kb < 8; ++kb)
    bw[kb] = *(const v16bf*)(WhT + (size_t)n * W_ + kb * 32 + half * 16);

  // h_0 = 0
  for (int i = threadIdx.x; i < 16 * W_; i += blockDim.x)
    ((__bf16*)Hb[0])[i] = f2bf(0.0f);
  __syncthreads();

  float hv[8];
#pragma unroll 1
  for (int t = 0; t < T_; ++t) {
    const int cur = t & 1, nxt = cur ^ 1;

    // Issue this step's xp loads first so they overlap the WMMA chain.
    float xv[8];
#pragma unroll
    for (int i = 0; i < 8; ++i)
      xv[i] = xp[((size_t)t * B_ + b0 + i + 8 * half) * W_ + n];

    // Prefetch the step-(t+4) tile into cache (global_prefetch_b8).
    if (t + 4 < T_)
      __builtin_prefetch(&xp[((size_t)(t + 4) * B_ + b0 + r) * W_ + wave * 16], 0, 0);

    v8f acc = {};
#pragma unroll
    for (int kb = 0; kb < 8; ++kb) {
      const int ka = kb * 32 + half * 16;
      v8bf l0 = *(const v8bf*)&Hb[cur][r][ka];        // ds_load_b128
      v8bf l1 = *(const v8bf*)&Hb[cur][r][ka + 8];
      v16bf a;
#pragma unroll
      for (int i = 0; i < 8; ++i) { a[i] = l0[i]; a[i + 8] = l1[i]; }
      acc = __builtin_amdgcn_wmma_f32_16x16x32_bf16(
          false, a, false, bw[kb], (short)0, acc, false, false);
    }

#pragma unroll
    for (int i = 0; i < 8; ++i) {
      hv[i] = fast_tanh(acc[i] + xv[i]);              // v_tanh_f32
      const __bf16 hb = f2bf(hv[i]);
      Hb[nxt][i + 8 * half][n] = hb;
      if (WRITE_SEQ)
        seq_out[((size_t)t * B_ + b0 + i + 8 * half) * W_ + n] = hb;
    }
    __syncthreads();   // publish h_{t+1} before next step's A-fragment reads
  }

  if (!WRITE_SEQ) {
#pragma unroll
    for (int i = 0; i < 8; ++i)
      h_out[(size_t)(b0 + i + 8 * half) * W_ + n] = hv[i];
  }
}

// ---------------------------------------------------------------------------
// out = softmax(h2 @ Wd + bd), C=2. Tiny: one thread per batch row.
// ---------------------------------------------------------------------------
__global__ void dense_softmax(const float* __restrict__ h,
                              const float* __restrict__ Wd,
                              const float* __restrict__ bd,
                              float* __restrict__ out) {
  int b = blockIdx.x * blockDim.x + threadIdx.x;
  if (b >= B_) return;
  float a0 = bd[0], a1 = bd[1];
  for (int k = 0; k < W_; ++k) {
    float hv = h[b * W_ + k];
    a0 = fmaf(hv, Wd[k * 2 + 0], a0);
    a1 = fmaf(hv, Wd[k * 2 + 1], a1);
  }
  float m = fmaxf(a0, a1);
  float e0 = expf(a0 - m), e1 = expf(a1 - m);
  float s = e0 + e1;
  out[b * 2 + 0] = e0 / s;
  out[b * 2 + 1] = e1 / s;
}

// ---------------------------------------------------------------------------
extern "C" void kernel_launch(void* const* d_in, const int* in_sizes, int n_in,
                              void* d_out, int out_size, void* d_ws, size_t ws_size,
                              hipStream_t stream) {
  (void)in_sizes; (void)n_in; (void)out_size; (void)ws_size;

  const int*   x   = (const int*)d_in[0];
  const float* emb = (const float*)d_in[1];
  const float* Wx1 = (const float*)d_in[2];
  const float* Wh1 = (const float*)d_in[3];
  const float* b1  = (const float*)d_in[4];
  const float* Wx2 = (const float*)d_in[5];
  const float* Wh2 = (const float*)d_in[6];
  const float* b2  = (const float*)d_in[7];
  const float* Wd  = (const float*)d_in[8];
  const float* bd  = (const float*)d_in[9];
  float* out = (float*)d_out;

  // Workspace carve-up (xp reused across both layers): ~201 MB total.
  char* p = (char*)d_ws;
  float*  xp   = (float*)p;  p += (size_t)B_ * T_ * W_ * sizeof(float);
  __bf16* s1   = (__bf16*)p; p += (size_t)B_ * T_ * W_ * sizeof(__bf16);
  __bf16* Wx1T = (__bf16*)p; p += (size_t)E_ * W_ * sizeof(__bf16);
  __bf16* Wh1T = (__bf16*)p; p += (size_t)W_ * W_ * sizeof(__bf16);
  __bf16* Wx2T = (__bf16*)p; p += (size_t)W_ * W_ * sizeof(__bf16);
  __bf16* Wh2T = (__bf16*)p; p += (size_t)W_ * W_ * sizeof(__bf16);
  float*  h2   = (float*)p;  p += (size_t)B_ * W_ * sizeof(float);

  // 1) Weight convert + transpose to bf16 WMMA-friendly layout.
  cvt_transpose<<<(E_ * W_ + 255) / 256, 256, 0, stream>>>(Wx1, Wx1T, E_, W_);
  cvt_transpose<<<(W_ * W_ + 255) / 256, 256, 0, stream>>>(Wh1, Wh1T, W_, W_);
  cvt_transpose<<<(W_ * W_ + 255) / 256, 256, 0, stream>>>(Wx2, Wx2T, W_, W_);
  cvt_transpose<<<(W_ * W_ + 255) / 256, 256, 0, stream>>>(Wh2, Wh2T, W_, W_);

  const int M = B_ * T_;
  const int gemm_blocks = (M / 16) * (W_ / 64) / 8;   // 4096

  // 2) xp1 = emb[x] @ Wx1 + b1  (bf16 WMMA, fp32 accumulate, time-major out)
  gemm_embed<<<gemm_blocks, 256, 0, stream>>>(x, emb, Wx1T, b1, xp);

  // 3) Layer-1 scan (batch-parallel across 4 blocks), emits s1 in bf16.
  rnn_scan<true><<<B_ / 16, 512, 0, stream>>>(xp, Wh1T, s1, nullptr);

  // 4) xp2 = s1 @ Wx2 + b2 (reuses xp buffer, time-major rows)
  gemm_bf16<<<gemm_blocks, 256, 0, stream>>>(s1, Wx2T, b2, xp, W_);

  // 5) Layer-2 scan, final state only.
  rnn_scan<false><<<B_ / 16, 512, 0, stream>>>(xp, Wh2T, nullptr, h2);

  // 6) Dense + softmax.
  dense_softmax<<<1, 64, 0, stream>>>(h2, Wd, bd, out);
}
